// MultiHeadAttention_5952824672331
// MI455X (gfx1250) — matrix operand order repaired and checked by reading, not compiled, not run on hardware
//
#include <hip/hip_runtime.h>

#define DMODEL 1472
#define INNER  384
#define SEQ    2048
#define BATCH  8
#define HEADS  6
#define DK     64
#define NTOK   (BATCH*SEQ)       // 16384
#define WELEMS (INNER*DMODEL)    // 565248 (Wq/Wk/Wv and Wo all have this many elements)

#ifndef __has_builtin
#define __has_builtin(x) 0
#endif

#if __has_builtin(__builtin_amdgcn_global_load_async_to_lds_b128)
#define HAVE_ASYNC 1
#else
#define HAVE_ASYNC 0
#endif

typedef __attribute__((ext_vector_type(16))) __bf16 v16bf;
typedef __attribute__((ext_vector_type(2)))  __bf16 v2bf;
typedef __attribute__((ext_vector_type(8)))  float  v8f;
typedef __attribute__((ext_vector_type(4)))  int    v4i;

union ABFrag { v16bf v; unsigned u[8]; };

// hardware cvt (v_cvt_pk_bf16_f32) instead of manual bit twiddling
static __device__ __forceinline__ unsigned short f2bf(float f) {
  return __builtin_bit_cast(unsigned short, (__bf16)f);
}
static __device__ __forceinline__ unsigned pack2(float a, float b) {
  v2bf v; v[0] = (__bf16)a; v[1] = (__bf16)b;
  return __builtin_bit_cast(unsigned, v);
}

// scheduling fence: keep prefetch issues above the compute block
static __device__ __forceinline__ void sched_fence() {
#if __has_builtin(__builtin_amdgcn_sched_barrier)
  __builtin_amdgcn_sched_barrier(0);
#endif
}

#if HAVE_ASYNC
static __device__ __forceinline__ void async_cp16(const void* g, void* l) {
  __builtin_amdgcn_global_load_async_to_lds_b128(
      (__attribute__((address_space(1))) v4i*)g,
      (__attribute__((address_space(3))) v4i*)l, 0, 0);
}
static __device__ __forceinline__ void wait_async0() {
#if __has_builtin(__builtin_amdgcn_s_wait_asynccnt)
  __builtin_amdgcn_s_wait_asynccnt(0);
#else
  asm volatile("s_wait_asynccnt 0" ::: "memory");
#endif
}
#endif

// A fragment: 16x32 bf16, lane m=lane&15 holds row m; K words per ISA layout:
// VGPR0..3 = K{0,1},{2,3},{4,5},{6,7}+8*kh ; VGPR4..7 = +16. kh = lane>>4.
static __device__ __forceinline__ void load_a_frag(ABFrag& f, const unsigned short* lds,
                                                   int row0, int ldw, int kbase) {
  const int lane = threadIdx.x & 31;
  const int m = lane & 15, kh = lane >> 4;
  const unsigned* p = (const unsigned*)(lds + (row0 + m) * ldw + kbase);
  const int w0 = kh * 4;
#pragma unroll
  for (int j = 0; j < 4; ++j) { f.u[j] = p[w0 + j]; f.u[4 + j] = p[w0 + 8 + j]; }
}

// B fragment: 32x16 bf16 (KxN), lane n=lane&15 holds column n; K words same as A fragment.
// LDS holds B column-major as [n][k] so each lane reads two runs of 4 contiguous K-pair words.
static __device__ __forceinline__ void load_b_frag(ABFrag& f, const unsigned short* lds,
                                                   int col0, int ldw, int kbase) {
  const int lane = threadIdx.x & 31;
  const int n = lane & 15, kh = lane >> 4;
  const unsigned* p = (const unsigned*)(lds + (col0 + n) * ldw + kbase);
  const int w0 = kh * 4;
#pragma unroll
  for (int j = 0; j < 4; ++j) { f.u[j] = p[w0 + j]; f.u[4 + j] = p[w0 + 8 + j]; }
}

static __device__ __forceinline__ v8f wmma_bf16(const ABFrag& a, const ABFrag& b, v8f c) {
  return __builtin_amdgcn_wmma_f32_16x16x32_bf16(false, a.v, false, b.v, (short)0, c,
                                                 false, false);
}

// ---------------------------------------------------------------------------
// Kernel 1: weights f32->bf16 + T5 relative-position bias table [H][4095]
// ---------------------------------------------------------------------------
__global__ __launch_bounds__(256) void prep_kernel(
    const float* __restrict__ Wq, const float* __restrict__ Wk,
    const float* __restrict__ Wv, const float* __restrict__ Wo,
    const float* __restrict__ rel_emb,
    unsigned short* __restrict__ wbf, float* __restrict__ biasTab)
{
  const int gid = blockIdx.x * 256 + threadIdx.x;
  if (gid < 4 * WELEMS) {
    const int which = gid / WELEMS, j = gid - which * WELEMS;
    const float* src = (which == 0) ? Wq : (which == 1) ? Wk : (which == 2) ? Wv : Wo;
    wbf[gid] = f2bf(src[j]);
  }
  if (gid < 4095) {
    const int rel = gid - 2047;               // mem - ctx
    const int ret = (rel > 0) ? 16 : 0;       // num_buckets//2 = 16
    const int n = rel < 0 ? -rel : rel;
    int bv;
    if (n < 8) bv = n;                        // max_exact = 8
    else {
      int t = 8 + (int)(__logf((float)n * 0.125f) * (8.0f / __logf(16.0f)));
      bv = t < 15 ? t : 15;
    }
    const int bucket = ret + bv;
#pragma unroll
    for (int h = 0; h < HEADS; ++h)
      biasTab[h * 4095 + gid] = rel_emb[bucket * HEADS + h];
  }
}

// ---------------------------------------------------------------------------
// Kernel 2: Q/K/V projection (y = x @ W^T), output bf16 in [B,H,S,DK] layout.
// Block: 256 thr = 8 waves, tile 128(M) x 128(N), wave tile 32x64, K step 32.
// Double-buffered: W tile via async DMA, X tile via float4 regs + cvt.
// ---------------------------------------------------------------------------
__global__ __launch_bounds__(256) void qkv_proj_kernel(
    const float* __restrict__ xq, const float* __restrict__ xkv,
    const unsigned short* __restrict__ wbf_base,
    unsigned short* __restrict__ qkv_base)
{
  __shared__ unsigned short Xs[2][128 * 40];
  __shared__ unsigned short Wsh[2][128 * 40];
  const int tid = threadIdx.x, lane = tid & 31, wave = tid >> 5;
  const int Mbase = blockIdx.x * 128, Nbase = blockIdx.y * 128;
  const int wsel = blockIdx.z;                          // 0=Q,1=K,2=V
  const float* __restrict__ x = (wsel == 0) ? xq : xkv;
  const unsigned short* __restrict__ w = wbf_base + (size_t)wsel * WELEMS;
  unsigned short* __restrict__ out = qkv_base + (size_t)wsel * ((size_t)NTOK * INNER);
  const int wm = (wave & 3) * 32, wn = (wave >> 2) * 64;
  const int NK = DMODEL / 32;                           // 46 k-steps
  v8f acc[2][4] = {};

  // ---- stage tile 0 into buffer 0 ----
  {
#if HAVE_ASYNC
#pragma unroll
    for (int j = 0; j < 2; ++j) {
      const int i = tid + j * 256, row = i >> 2, c8 = (i & 3) * 8;
      async_cp16(w + (size_t)(Nbase + row) * DMODEL + c8, &Wsh[0][row * 40 + c8]);
    }
#else
#pragma unroll
    for (int j = 0; j < 8; ++j) {
      const int i = tid + j * 256, row = i >> 4, c2 = (i & 15) * 2;
      *(unsigned*)&Wsh[0][row * 40 + c2] =
          *(const unsigned*)(w + (size_t)(Nbase + row) * DMODEL + c2);
    }
#endif
#pragma unroll
    for (int j = 0; j < 4; ++j) {
      const int i = tid + j * 256, row = i >> 3, c4 = (i & 7) * 4;
      const float4 f4 = *(const float4*)(x + (size_t)(Mbase + row) * DMODEL + c4);
      unsigned* dst = (unsigned*)&Xs[0][row * 40 + c4];
      dst[0] = pack2(f4.x, f4.y);
      dst[1] = pack2(f4.z, f4.w);
    }
#if HAVE_ASYNC
    wait_async0();
#endif
    __syncthreads();
  }

  for (int kk = 0; kk < NK; ++kk) {
    const int cur = kk & 1, nxt = cur ^ 1;
    const bool pre = (kk + 1 < NK);
    float4 xr[4];
    if (pre) {                                          // prefetch tile kk+1
      const int k0 = (kk + 1) * 32;
#if HAVE_ASYNC
#pragma unroll
      for (int j = 0; j < 2; ++j) {
        const int i = tid + j * 256, row = i >> 2, c8 = (i & 3) * 8;
        async_cp16(w + (size_t)(Nbase + row) * DMODEL + k0 + c8,
                   &Wsh[nxt][row * 40 + c8]);
      }
#else
#pragma unroll
      for (int j = 0; j < 8; ++j) {
        const int i = tid + j * 256, row = i >> 4, c2 = (i & 15) * 2;
        *(unsigned*)&Wsh[nxt][row * 40 + c2] =
            *(const unsigned*)(w + (size_t)(Nbase + row) * DMODEL + k0 + c2);
      }
#endif
#pragma unroll
      for (int j = 0; j < 4; ++j) {
        const int i = tid + j * 256, row = i >> 3, c4 = (i & 7) * 4;
        xr[j] = *(const float4*)(x + (size_t)(Mbase + row) * DMODEL + k0 + c4);
      }
    }
    sched_fence();   // pin all prefetch issues above the WMMA block
    // ---- compute on current buffer (overlaps with DMA + loads of next tile) ----
    ABFrag a0, a1;
    load_a_frag(a0, Xs[cur], wm, 40, 0);
    load_a_frag(a1, Xs[cur], wm + 16, 40, 0);
#pragma unroll
    for (int in = 0; in < 4; ++in) {
      ABFrag b;
      load_b_frag(b, Wsh[cur], wn + in * 16, 40, 0);
      acc[0][in] = wmma_bf16(a0, b, acc[0][in]);
      acc[1][in] = wmma_bf16(a1, b, acc[1][in]);
    }
    if (pre) {                                          // cvt + store X(kk+1)
#pragma unroll
      for (int j = 0; j < 4; ++j) {
        const int i = tid + j * 256, row = i >> 3, c4 = (i & 7) * 4;
        unsigned* dst = (unsigned*)&Xs[nxt][row * 40 + c4];
        dst[0] = pack2(xr[j].x, xr[j].y);
        dst[1] = pack2(xr[j].z, xr[j].w);
      }
    }
#if HAVE_ASYNC
    wait_async0();
#endif
    __syncthreads();
  }
  // epilogue: C layout row = r + 8*(lane>>4), col = lane&15
#pragma unroll
  for (int im = 0; im < 2; ++im)
#pragma unroll
    for (int in = 0; in < 4; ++in) {
      const int nglob = Nbase + wn + in * 16 + (lane & 15);
      const int h = nglob >> 6, d = nglob & 63;
#pragma unroll
      for (int r = 0; r < 8; ++r) {
        const int m = Mbase + wm + im * 16 + r + 8 * (lane >> 4);
        const int b = m >> 11, s = m & 2047;
        const size_t addr = ((size_t)(b * HEADS + h) << 17) | ((size_t)s << 6) | (size_t)d;
        out[addr] = f2bf(acc[im][in][r]);
      }
    }
}

// ---------------------------------------------------------------------------
// Kernel 3: flash attention, 4 waves, 64 q-rows per block, KV tiles of 64.
// scores = Q K^T + bias(kv-q); online softmax; O += P V.
// Double-buffered K (async DMA), V (regs -> transposed store), bias slice.
// ---------------------------------------------------------------------------
__global__ __launch_bounds__(128) void attn_kernel(
    const unsigned short* __restrict__ Qbf, const unsigned short* __restrict__ Kbf,
    const unsigned short* __restrict__ Vbf, unsigned short* __restrict__ Obf,
    const float* __restrict__ biasTab)
{
  __shared__ unsigned short Qs[64 * 72];
  __shared__ unsigned short Ks[2][64 * 72];
  __shared__ unsigned short Vts[2][64 * 72];  // V transposed: [d][kv]
  __shared__ unsigned short Ps[4 * 16 * 72];  // per-wave P scratch (C->A relayout)
  __shared__ float biasS[2][128];
  const int tid = threadIdx.x, lane = tid & 31, wave = tid >> 5;
  const int q0 = blockIdx.x * 64, h = blockIdx.y, b = blockIdx.z;
  const size_t bh = ((size_t)(b * HEADS + h)) << 17;    // *S*DK
  const int NT = SEQ / 64;                              // 32 kv tiles

  // ---- prologue: Q tile + KV tile 0 into buffer 0 ----
#if HAVE_ASYNC
#pragma unroll
  for (int j = 0; j < 4; ++j) {
    const int i = tid + j * 128, row = i >> 3, c8 = (i & 7) * 8;
    async_cp16(Qbf + bh + ((size_t)(q0 + row) << 6) + c8, &Qs[row * 72 + c8]);
    async_cp16(Kbf + bh + ((size_t)row << 6) + c8, &Ks[0][row * 72 + c8]);
  }
#else
#pragma unroll
  for (int j = 0; j < 16; ++j) {
    const int i = tid + j * 128, row = i >> 5, c2 = (i & 31) * 2;
    *(unsigned*)&Qs[row * 72 + c2] =
        *(const unsigned*)(Qbf + bh + ((size_t)(q0 + row) << 6) + c2);
    *(unsigned*)&Ks[0][row * 72 + c2] =
        *(const unsigned*)(Kbf + bh + ((size_t)row << 6) + c2);
  }
#endif
#pragma unroll
  for (int j = 0; j < 16; ++j) {                        // V tile 0, transposed
    const int i = tid + j * 128, row = i >> 5, dd = (i & 31) * 2;
    const unsigned u = *(const unsigned*)(Vbf + bh + ((size_t)row << 6) + dd);
    Vts[0][dd * 72 + row] = (unsigned short)(u & 0xffffu);
    Vts[0][(dd + 1) * 72 + row] = (unsigned short)(u >> 16);
  }
  if (tid < 127)
    biasS[0][tid] = biasTab[h * 4095 + (0 - q0) + (tid - 63) + 2047];
#if HAVE_ASYNC
  wait_async0();
#endif

  float mrow[8], lrow[8];
  v8f oacc[4] = {};
#pragma unroll
  for (int r = 0; r < 8; ++r) { mrow[r] = -1e30f; lrow[r] = 0.0f; }
  __syncthreads();

  unsigned short* pw = &Ps[wave * 16 * 72];
  for (int kt = 0; kt < NT; ++kt) {
    const int cur = kt & 1, nxt = cur ^ 1;
    const bool pre = (kt + 1 < NT);
    unsigned vreg[16];
#if !HAVE_ASYNC
    unsigned kreg[16];
#endif
    float breg = 0.0f;
    if (pre) {                                          // prefetch tile kt+1
      const int kvn = (kt + 1) * 64;
#if HAVE_ASYNC
#pragma unroll
      for (int j = 0; j < 4; ++j) {
        const int i = tid + j * 128, row = i >> 3, c8 = (i & 7) * 8;
        async_cp16(Kbf + bh + ((size_t)(kvn + row) << 6) + c8,
                   &Ks[nxt][row * 72 + c8]);
      }
#else
#pragma unroll
      for (int j = 0; j < 16; ++j) {
        const int i = tid + j * 128, row = i >> 5, c2 = (i & 31) * 2;
        kreg[j] = *(const unsigned*)(Kbf + bh + ((size_t)(kvn + row) << 6) + c2);
      }
#endif
#pragma unroll
      for (int j = 0; j < 16; ++j) {
        const int i = tid + j * 128, row = i >> 5, dd = (i & 31) * 2;
        vreg[j] = *(const unsigned*)(Vbf + bh + ((size_t)(kvn + row) << 6) + dd);
      }
      if (tid < 127)
        breg = biasTab[h * 4095 + (kvn - q0) + (tid - 63) + 2047];
    }
    sched_fence();   // pin prefetch issues above the compute block

    // ---- scores: 16 q-rows x 64 kv-cols per wave ----
    v8f sc[4] = {};
    ABFrag aq0, aq1;
    load_a_frag(aq0, Qs, wave * 16, 72, 0);
    load_a_frag(aq1, Qs, wave * 16, 72, 32);
#pragma unroll
    for (int nt = 0; nt < 4; ++nt) {
      ABFrag bk;
      load_b_frag(bk, Ks[cur], nt * 16, 72, 0);
      sc[nt] = wmma_bf16(aq0, bk, sc[nt]);
      load_b_frag(bk, Ks[cur], nt * 16, 72, 32);
      sc[nt] = wmma_bf16(aq1, bk, sc[nt]);
    }

    // bias + per-row max (rows live in 16-lane halves of the C fragment)
    const int qrow_base = wave * 16 + 8 * (lane >> 4);
    float rmax[8];
#pragma unroll
    for (int r = 0; r < 8; ++r) rmax[r] = -1e30f;
#pragma unroll
    for (int nt = 0; nt < 4; ++nt) {
      const int kvl = nt * 16 + (lane & 15);
#pragma unroll
      for (int r = 0; r < 8; ++r) {
        const float v = sc[nt][r] + biasS[cur][kvl - (qrow_base + r) + 63];
        sc[nt][r] = v;
        rmax[r] = fmaxf(rmax[r], v);
      }
    }
    float scale[8], rsum[8];
#pragma unroll
    for (int r = 0; r < 8; ++r) {
#pragma unroll
      for (int msk = 1; msk < 16; msk <<= 1)
        rmax[r] = fmaxf(rmax[r], __shfl_xor(rmax[r], msk, 32));
      const float mnew = fmaxf(mrow[r], rmax[r]);
      scale[r] = __expf(mrow[r] - mnew);
      mrow[r] = mnew;
      rsum[r] = 0.0f;
    }
#pragma unroll
    for (int nt = 0; nt < 4; ++nt)
#pragma unroll
      for (int r = 0; r < 8; ++r) {
        const float p = __expf(sc[nt][r] - mrow[r]);
        sc[nt][r] = p;
        rsum[r] += p;
      }
#pragma unroll
    for (int r = 0; r < 8; ++r) {
#pragma unroll
      for (int msk = 1; msk < 16; msk <<= 1) rsum[r] += __shfl_xor(rsum[r], msk, 32);
      lrow[r] = lrow[r] * scale[r] + rsum[r];
    }
#pragma unroll
    for (int nt = 0; nt < 4; ++nt)
#pragma unroll
      for (int r = 0; r < 8; ++r) oacc[nt][r] *= scale[r];

    // P (C layout) -> LDS -> A fragment, then P @ V
#pragma unroll
    for (int nt = 0; nt < 4; ++nt) {
      const int kvl = nt * 16 + (lane & 15);
#pragma unroll
      for (int r = 0; r < 8; ++r)
        pw[(r + 8 * (lane >> 4)) * 72 + kvl] = f2bf(sc[nt][r]);
    }
    ABFrag ap0, ap1;
    load_a_frag(ap0, pw, 0, 72, 0);
    load_a_frag(ap1, pw, 0, 72, 32);
#pragma unroll
    for (int nt = 0; nt < 4; ++nt) {
      ABFrag bv;
      load_b_frag(bv, Vts[cur], nt * 16, 72, 0);
      oacc[nt] = wmma_bf16(ap0, bv, oacc[nt]);
      load_b_frag(bv, Vts[cur], nt * 16, 72, 32);
      oacc[nt] = wmma_bf16(ap1, bv, oacc[nt]);
    }

    if (pre) {                                          // commit staged tile kt+1
#if !HAVE_ASYNC
#pragma unroll
      for (int j = 0; j < 16; ++j) {
        const int i = tid + j * 128, row = i >> 5, c2 = (i & 31) * 2;
        *(unsigned*)&Ks[nxt][row * 72 + c2] = kreg[j];
      }
#endif
#pragma unroll
      for (int j = 0; j < 16; ++j) {
        const int i = tid + j * 128, row = i >> 5, dd = (i & 31) * 2;
        Vts[nxt][dd * 72 + row] = (unsigned short)(vreg[j] & 0xffffu);
        Vts[nxt][(dd + 1) * 72 + row] = (unsigned short)(vreg[j] >> 16);
      }
      if (tid < 127) biasS[nxt][tid] = breg;
    }
#if HAVE_ASYNC
    wait_async0();
#endif
    __syncthreads();
  }

  // normalize + store attn-out bf16 in token-major [B*S, INNER]
#pragma unroll
  for (int nt = 0; nt < 4; ++nt) {
    const int d = h * DK + nt * 16 + (lane & 15);
#pragma unroll
    for (int r = 0; r < 8; ++r) {
      const int qq = q0 + wave * 16 + r + 8 * (lane >> 4);
      const size_t t = (size_t)b * SEQ + qq;
      Obf[t * INNER + d] = f2bf(oacc[nt][r] / lrow[r]);
    }
  }
}

// ---------------------------------------------------------------------------
// Kernel 4: output projection  out = O @ Wo^T  (f32 result)
// Block: 256 thr, tile 128(M) x 64(N), wave tile 32x32 (N=1472 = 23*64).
// Double-buffered: both tiles via async DMA (bf16 sources).
// ---------------------------------------------------------------------------
__global__ __launch_bounds__(256) void oproj_kernel(
    const unsigned short* __restrict__ Obf, const unsigned short* __restrict__ WoBf,
    float* __restrict__ out)
{
  __shared__ unsigned short As[2][128 * 40];
  __shared__ unsigned short Bs[2][64 * 40];
  const int tid = threadIdx.x, lane = tid & 31, wave = tid >> 5;
  const int Mbase = blockIdx.x * 128, Nbase = blockIdx.y * 64;
  const int wm = (wave & 3) * 32, wn = (wave >> 2) * 32;
  const int NK = INNER / 32;                            // 12 k-steps
  v8f acc[2][2] = {};

  // ---- stage tile 0 ----
#if HAVE_ASYNC
#pragma unroll
  for (int j = 0; j < 2; ++j) {
    const int i = tid + j * 256, row = i >> 2, c8 = (i & 3) * 8;
    async_cp16(Obf + (size_t)(Mbase + row) * INNER + c8, &As[0][row * 40 + c8]);
  }
  {
    const int row = tid >> 2, c8 = (tid & 3) * 8;
    async_cp16(WoBf + (size_t)(Nbase + row) * INNER + c8, &Bs[0][row * 40 + c8]);
  }
  wait_async0();
#else
#pragma unroll
  for (int j = 0; j < 8; ++j) {
    const int i = tid + j * 256, row = i >> 4, c2 = (i & 15) * 2;
    *(unsigned*)&As[0][row * 40 + c2] =
        *(const unsigned*)(Obf + (size_t)(Mbase + row) * INNER + c2);
  }
#pragma unroll
  for (int j = 0; j < 4; ++j) {
    const int i = tid + j * 256, row = i >> 4, c2 = (i & 15) * 2;
    *(unsigned*)&Bs[0][row * 40 + c2] =
        *(const unsigned*)(WoBf + (size_t)(Nbase + row) * INNER + c2);
  }
#endif
  __syncthreads();

  for (int kk = 0; kk < NK; ++kk) {
    const int cur = kk & 1, nxt = cur ^ 1;
    const bool pre = (kk + 1 < NK);
    if (pre) {                                          // prefetch tile kk+1
      const int k0 = (kk + 1) * 32;
#if HAVE_ASYNC
#pragma unroll
      for (int j = 0; j < 2; ++j) {
        const int i = tid + j * 256, row = i >> 2, c8 = (i & 3) * 8;
        async_cp16(Obf + (size_t)(Mbase + row) * INNER + k0 + c8,
                   &As[nxt][row * 40 + c8]);
      }
      {
        const int row = tid >> 2, c8 = (tid & 3) * 8;
        async_cp16(WoBf + (size_t)(Nbase + row) * INNER + k0 + c8,
                   &Bs[nxt][row * 40 + c8]);
      }
#else
#pragma unroll
      for (int j = 0; j < 8; ++j) {
        const int i = tid + j * 256, row = i >> 4, c2 = (i & 15) * 2;
        *(unsigned*)&As[nxt][row * 40 + c2] =
            *(const unsigned*)(Obf + (size_t)(Mbase + row) * INNER + k0 + c2);
      }
#pragma unroll
      for (int j = 0; j < 4; ++j) {
        const int i = tid + j * 256, row = i >> 4, c2 = (i & 15) * 2;
        *(unsigned*)&Bs[nxt][row * 40 + c2] =
            *(const unsigned*)(WoBf + (size_t)(Nbase + row) * INNER + k0 + c2);
      }
#endif
    }
    sched_fence();   // pin prefetch issues above the WMMA block
    ABFrag a0, a1, b0, b1;
    load_a_frag(a0, As[cur], wm, 40, 0);
    load_a_frag(a1, As[cur], wm + 16, 40, 0);
    load_b_frag(b0, Bs[cur], wn, 40, 0);
    load_b_frag(b1, Bs[cur], wn + 16, 40, 0);
    acc[0][0] = wmma_bf16(a0, b0, acc[0][0]);
    acc[0][1] = wmma_bf16(a0, b1, acc[0][1]);
    acc[1][0] = wmma_bf16(a1, b0, acc[1][0]);
    acc[1][1] = wmma_bf16(a1, b1, acc[1][1]);
#if HAVE_ASYNC
    wait_async0();
#endif
    __syncthreads();
  }
#pragma unroll
  for (int im = 0; im < 2; ++im)
#pragma unroll
    for (int in = 0; in < 2; ++in) {
      const int n = Nbase + wn + in * 16 + (lane & 15);
#pragma unroll
      for (int r = 0; r < 8; ++r) {
        const int m = Mbase + wm + im * 16 + r + 8 * (lane >> 4);
        out[(size_t)m * DMODEL + n] = acc[im][in][r];
      }
    }
}

// ---------------------------------------------------------------------------
extern "C" void kernel_launch(void* const* d_in, const int* in_sizes, int n_in,
                              void* d_out, int out_size, void* d_ws, size_t ws_size,
                              hipStream_t stream) {
  (void)in_sizes; (void)n_in; (void)out_size; (void)ws_size;
  const float* xq  = (const float*)d_in[0];
  const float* xkv = (const float*)d_in[1];
  const float* Wq  = (const float*)d_in[2];
  const float* Wk  = (const float*)d_in[3];
  const float* Wv  = (const float*)d_in[4];
  const float* Wo  = (const float*)d_in[5];
  const float* rel = (const float*)d_in[6];

  // workspace layout (bytes): Q,K,V,O bf16 activations; bf16 weights; bias table
  char* ws = (char*)d_ws;
  const size_t SZT = (size_t)NTOK * INNER * 2;                 // 12.58 MB each
  unsigned short* Qbf = (unsigned short*)(ws);
  unsigned short* Kbf = (unsigned short*)(ws + SZT);
  unsigned short* Vbf = (unsigned short*)(ws + 2 * SZT);
  unsigned short* Obf = (unsigned short*)(ws + 3 * SZT);
  unsigned short* wbf = (unsigned short*)(ws + 4 * SZT);       // Wq,Wk,Wv,Wo bf16
  float* biasTab      = (float*)(ws + 4 * SZT + (size_t)4 * WELEMS * 2);

  prep_kernel<<<dim3((4 * WELEMS + 255) / 256), dim3(256), 0, stream>>>(
      Wq, Wk, Wv, Wo, rel, wbf, biasTab);
  qkv_proj_kernel<<<dim3(NTOK / 128, INNER / 128, 3), dim3(256), 0, stream>>>(
      xq, xkv, wbf, Qbf);
  attn_kernel<<<dim3(SEQ / 64, HEADS, BATCH), dim3(128), 0, stream>>>(
      Qbf, Kbf, Vbf, Obf, biasTab);
  oproj_kernel<<<dim3(NTOK / 128, DMODEL / 64), dim3(256), 0, stream>>>(
      Obf, wbf + (size_t)3 * WELEMS, (float*)d_out);
}
